// PerformerAttention_19542101197412
// MI455X (gfx1250) — compile-verified, matrix operand order checked by reading, not run on hardware
//
#include <hip/hip_runtime.h>
#include <hip/hip_bf16.h>

typedef __bf16 bf16_t;
typedef __attribute__((ext_vector_type(16))) __bf16 v16bf;
typedef __attribute__((ext_vector_type(8)))  __bf16 v8bf;
typedef __attribute__((ext_vector_type(8)))  float  v8f;

#define HEADS  8
#define DHEAD  64
#define NSEQ   4096
#define BATCH  2
#define DMODEL 512
#define EQKV   1536
#define MTOT   (BATCH * NSEQ) /* 8192 rows */

// ---------------------------------------------------------------------------
// D = A*B + C, A/B bf16 16x32 / 32x16, C/D f32 16x16, wave32.
static __device__ inline v8f wmma_bf16(v16bf a, v16bf b, v8f c) {
  return __builtin_amdgcn_wmma_f32_16x16x32_bf16(
      /*neg_a=*/false, a, /*neg_b=*/false, b,
      /*c_mod=*/(short)0, c, /*reuse_a=*/false, /*reuse_b=*/false);
}

// Load one 16x32 WMMA operand tile from a row-major bf16 matrix (global or
// LDS): two 16-byte vector loads per lane.  A/B operand layouts are
// symmetric: lane L holds row (L&15); khalf = (L>>4)*8;
// elements 0..7 <- K=khalf+0..7, elements 8..15 <- K=khalf+16..23.
static __device__ inline v16bf load_op_bf16(const bf16_t* p, int ld, int lane) {
  const bf16_t* r = p + (size_t)(lane & 15) * ld + ((lane >> 4) << 3);
  v8bf lo = *(const v8bf*)r;
  v8bf hi = *(const v8bf*)(r + 16);
  v16bf o;
#pragma unroll
  for (int i = 0; i < 8; ++i) { o[i] = lo[i]; o[8 + i] = hi[i]; }
  return o;
}

// ---------------------------------------------------------------------------
// Kernel 0: one-shot f32 -> bf16 conversion (8 elements/thread).
__global__ __launch_bounds__(256) void cvt_f32_bf16_kernel(
    const float* __restrict__ src, bf16_t* __restrict__ dst, int n) {
  int i = (blockIdx.x * 256 + threadIdx.x) * 8;
  if (i >= n) return;
  const float4 a = *(const float4*)(src + i);
  const float4 b = *(const float4*)(src + i + 4);
  v8bf o;
  o[0] = (__bf16)a.x; o[1] = (__bf16)a.y; o[2] = (__bf16)a.z; o[3] = (__bf16)a.w;
  o[4] = (__bf16)b.x; o[5] = (__bf16)b.y; o[6] = (__bf16)b.z; o[7] = (__bf16)b.w;
  *(v8bf*)(dst + i) = o;
}

// ---------------------------------------------------------------------------
// Kernel 1: qkv = x @ Wqkv^T (M=8192, K=512, E=1536), bf16 inputs, scattered
// into Q (scaled 1/8, [b,h,n,d]), K ([b,h,n,d]), V transposed ([b,h,d,n]).
// One wave computes a 32x64 tile: every B tile feeds two WMMAs.
__global__ __launch_bounds__(256) void qkv_proj_kernel(
    const bf16_t* __restrict__ x, const bf16_t* __restrict__ Wqkv,
    bf16_t* __restrict__ Q, bf16_t* __restrict__ K, bf16_t* __restrict__ Vt) {
  const int lane = threadIdx.x & 31;
  const int wid  = blockIdx.x * (blockDim.x >> 5) + (threadIdx.x >> 5);
  const int ET = EQKV / 64;                 // 24 column tiles
  const int et = wid % ET;
  const int mt = wid / ET;
  const int m0 = mt * 32;
  const int e0 = et * 64;

  v8f acc0[4] = {}, acc1[4] = {};
  for (int k0 = 0; k0 < DMODEL; k0 += 32) {
    v16bf a0 = load_op_bf16(x + (size_t)m0 * DMODEL + k0, DMODEL, lane);
    v16bf a1 = load_op_bf16(x + (size_t)(m0 + 16) * DMODEL + k0, DMODEL, lane);
#pragma unroll
    for (int j = 0; j < 4; ++j) {
      v16bf b = load_op_bf16(Wqkv + (size_t)(e0 + j * 16) * DMODEL + k0, DMODEL, lane);
      acc0[j] = wmma_bf16(a0, b, acc0[j]);
      acc1[j] = wmma_bf16(a1, b, acc1[j]);
    }
  }

  // C/D layout: lane L -> column n = L&15; VGPR r -> row m = r + (L<16?0:8).
  const int mo = (lane < 16) ? 0 : 8;
  const int nc = lane & 15;
#pragma unroll
  for (int mi = 0; mi < 2; ++mi) {
    const v8f* acc = mi ? acc1 : acc0;
#pragma unroll
    for (int j = 0; j < 4; ++j) {
#pragma unroll
      for (int r = 0; r < 8; ++r) {
        float v = acc[j][r];
        int m = m0 + mi * 16 + mo + r;
        int e = e0 + j * 16 + nc;
        int b    = m >> 12;                 // / NSEQ
        int n    = m & (NSEQ - 1);
        int part = e >> 9;                  // 0=q 1=k 2=v
        int rem  = e & 511;
        int h    = rem >> 6;
        int d    = rem & 63;
        int bh   = b * HEADS + h;
        if (part == 0)      Q [((size_t)bh * NSEQ + n) * DHEAD + d] = (__bf16)(v * 0.125f);
        else if (part == 1) K [((size_t)bh * NSEQ + n) * DHEAD + d] = (__bf16)v;
        else                Vt[((size_t)bh * DHEAD + d) * NSEQ + n] = (__bf16)v;
      }
    }
  }
}

// ---------------------------------------------------------------------------
// Kernel 2: streaming ReLU attention.  One block = (b,h) x 256 query rows;
// each of the 8 waves owns 32 query rows (two 16-row tiles) and two f32
// 16x64 accumulators.  Per 32-key step: 8 WMMA for S = Q*K^T (each K tile
// reused for both query tiles), ReLU, bf16 restage via wave-private LDS
// (C-layout -> A-layout), 8 WMMA for acc += S*V (each V tile reused twice).
__global__ __launch_bounds__(256) void relu_attn_kernel(
    const bf16_t* __restrict__ Q, const bf16_t* __restrict__ K,
    const bf16_t* __restrict__ Vt, bf16_t* __restrict__ attn) {
  __shared__ __align__(16) bf16_t s_stage[8][2][16 * 32];  // 16 KB, wave-private

  const int lane = threadIdx.x & 31;
  const int w    = threadIdx.x >> 5;
  const int NT   = NSEQ / 256;              // 16 query blocks per (b,h)
  const int bh   = blockIdx.x / NT;
  const int qt   = blockIdx.x % NT;
  const int q0   = qt * 256 + w * 32;

  const bf16_t* Qb = Q  + (size_t)bh * NSEQ * DHEAD;
  const bf16_t* Kb = K  + (size_t)bh * NSEQ * DHEAD;
  const bf16_t* Vb = Vt + (size_t)bh * DHEAD * NSEQ;
  bf16_t* sw0 = s_stage[w][0];
  bf16_t* sw1 = s_stage[w][1];

  // Q rows for this wave stay resident in registers (2 tiles x d-halves).
  const v16bf qa00 = load_op_bf16(Qb + (size_t)q0 * DHEAD + 0,  DHEAD, lane);
  const v16bf qa01 = load_op_bf16(Qb + (size_t)q0 * DHEAD + 32, DHEAD, lane);
  const v16bf qa10 = load_op_bf16(Qb + (size_t)(q0 + 16) * DHEAD + 0,  DHEAD, lane);
  const v16bf qa11 = load_op_bf16(Qb + (size_t)(q0 + 16) * DHEAD + 32, DHEAD, lane);

  v8f acc0[4] = {}, acc1[4] = {};
  const int nc = lane & 15;
  const int mo = (lane < 16) ? 0 : 8;

  for (int kt = 0; kt < NSEQ; kt += 32) {
    // Prefetch the next K/V tiles while this step computes.
    if (kt + 32 < NSEQ) {
      __builtin_prefetch(Kb + (size_t)(kt + 32 + lane) * DHEAD, 0, 0);
      __builtin_prefetch(Vb + (size_t)lane * NSEQ + kt + 32, 0, 0);
      __builtin_prefetch(Vb + (size_t)(lane + 32) * NSEQ + kt + 32, 0, 0);
    }

    // S = (Q*scale) @ K^T for 2x16 queries x 32 keys, f32 accum.
    v8f s00 = {}, s01 = {}, s10 = {}, s11 = {};
    {
      v16bf kb;
      kb = load_op_bf16(Kb + (size_t)kt * DHEAD + 0, DHEAD, lane);
      s00 = wmma_bf16(qa00, kb, s00);
      s10 = wmma_bf16(qa10, kb, s10);
      kb = load_op_bf16(Kb + (size_t)kt * DHEAD + 32, DHEAD, lane);
      s00 = wmma_bf16(qa01, kb, s00);
      s10 = wmma_bf16(qa11, kb, s10);
      kb = load_op_bf16(Kb + (size_t)(kt + 16) * DHEAD + 0, DHEAD, lane);
      s01 = wmma_bf16(qa00, kb, s01);
      s11 = wmma_bf16(qa10, kb, s11);
      kb = load_op_bf16(Kb + (size_t)(kt + 16) * DHEAD + 32, DHEAD, lane);
      s01 = wmma_bf16(qa01, kb, s01);
      s11 = wmma_bf16(qa11, kb, s11);
    }

    // ReLU + restage as row-major bf16 16x32 tiles in this wave's LDS.
    // (wave-private: DS ops from one wave are in-order, no barrier needed)
#pragma unroll
    for (int r = 0; r < 8; ++r) {
      sw0[(mo + r) * 32 + nc]      = (__bf16)fmaxf(s00[r], 0.0f);
      sw0[(mo + r) * 32 + nc + 16] = (__bf16)fmaxf(s01[r], 0.0f);
      sw1[(mo + r) * 32 + nc]      = (__bf16)fmaxf(s10[r], 0.0f);
      sw1[(mo + r) * 32 + nc + 16] = (__bf16)fmaxf(s11[r], 0.0f);
    }

    // Reload S in A-operand layout; V columns are rows of Vt (contiguous).
    v16bf sa0 = load_op_bf16(sw0, 32, lane);
    v16bf sa1 = load_op_bf16(sw1, 32, lane);
#pragma unroll
    for (int j = 0; j < 4; ++j) {
      v16bf vb = load_op_bf16(Vb + (size_t)(j * 16) * NSEQ + kt, NSEQ, lane);
      acc0[j] = wmma_bf16(sa0, vb, acc0[j]);
      acc1[j] = wmma_bf16(sa1, vb, acc1[j]);
    }
  }

  // Write per-head output back in [b, n, h*64+d] (bf16) for the final GEMM.
  const int b = bh / HEADS, h = bh % HEADS;
#pragma unroll
  for (int mi = 0; mi < 2; ++mi) {
    const v8f* acc = mi ? acc1 : acc0;
#pragma unroll
    for (int j = 0; j < 4; ++j)
#pragma unroll
      for (int r = 0; r < 8; ++r)
        attn[((size_t)(b * NSEQ + q0 + mi * 16 + mo + r)) * DMODEL + h * DHEAD + j * 16 + nc] =
            (__bf16)acc[j][r];
  }
}

// ---------------------------------------------------------------------------
// Kernel 3: out = attn @ Wout^T + b_out (M=8192, K=512, E=512), f32 output.
// One wave computes a 32x64 tile, pure bf16 operands.
__global__ __launch_bounds__(256) void out_proj_kernel(
    const bf16_t* __restrict__ attn, const bf16_t* __restrict__ Wout,
    const float* __restrict__ bias, float* __restrict__ out) {
  const int lane = threadIdx.x & 31;
  const int wid  = blockIdx.x * (blockDim.x >> 5) + (threadIdx.x >> 5);
  const int ET = DMODEL / 64;               // 8 column tiles
  const int et = wid % ET;
  const int mt = wid / ET;
  const int m0 = mt * 32, e0 = et * 64;

  v8f acc0[4] = {}, acc1[4] = {};
  for (int k0 = 0; k0 < DMODEL; k0 += 32) {
    v16bf a0 = load_op_bf16(attn + (size_t)m0 * DMODEL + k0, DMODEL, lane);
    v16bf a1 = load_op_bf16(attn + (size_t)(m0 + 16) * DMODEL + k0, DMODEL, lane);
#pragma unroll
    for (int j = 0; j < 4; ++j) {
      v16bf b = load_op_bf16(Wout + (size_t)(e0 + j * 16) * DMODEL + k0, DMODEL, lane);
      acc0[j] = wmma_bf16(a0, b, acc0[j]);
      acc1[j] = wmma_bf16(a1, b, acc1[j]);
    }
  }

  const int mo = (lane < 16) ? 0 : 8;
  const int nc = lane & 15;
#pragma unroll
  for (int mi = 0; mi < 2; ++mi) {
    const v8f* acc = mi ? acc1 : acc0;
#pragma unroll
    for (int j = 0; j < 4; ++j)
#pragma unroll
      for (int r = 0; r < 8; ++r) {
        int e = e0 + j * 16 + nc;
        out[(size_t)(m0 + mi * 16 + mo + r) * DMODEL + e] = acc[j][r] + bias[e];
      }
  }
}

// ---------------------------------------------------------------------------
extern "C" void kernel_launch(void* const* d_in, const int* in_sizes, int n_in,
                              void* d_out, int out_size, void* d_ws, size_t ws_size,
                              hipStream_t stream) {
  (void)in_sizes; (void)n_in; (void)out_size; (void)ws_size;
  const float* x    = (const float*)d_in[0];
  const float* Wqkv = (const float*)d_in[1];
  const float* Wout = (const float*)d_in[2];
  const float* bias = (const float*)d_in[3];
  float* out = (float*)d_out;

  // Workspace layout (bf16):
  //   Q | K | Vt | attn   (4 x 4,194,304 elems)
  //   xb (4,194,304) | Wqkvb (786,432) | Woutb (262,144)   => ~44 MB total
  const size_t per   = (size_t)BATCH * HEADS * NSEQ * DHEAD;
  const int n_x      = MTOT * DMODEL;      // 4,194,304
  const int n_wqkv   = EQKV * DMODEL;      //   786,432
  const int n_wout   = DMODEL * DMODEL;    //   262,144
  bf16_t* Q     = (bf16_t*)d_ws;
  bf16_t* K     = Q + per;
  bf16_t* Vt    = K + per;
  bf16_t* attn  = Vt + per;
  bf16_t* xb    = attn + per;
  bf16_t* Wqkvb = xb + n_x;
  bf16_t* Woutb = Wqkvb + n_wqkv;

  // One-shot f32 -> bf16 conversion (each element converted exactly once).
  cvt_f32_bf16_kernel<<<n_x    / (8 * 256), 256, 0, stream>>>(x,    xb,    n_x);
  cvt_f32_bf16_kernel<<<n_wqkv / (8 * 256), 256, 0, stream>>>(Wqkv, Wqkvb, n_wqkv);
  cvt_f32_bf16_kernel<<<n_wout / (8 * 256), 256, 0, stream>>>(Wout, Woutb, n_wout);

  {
    int tiles = (MTOT / 32) * (EQKV / 64);  // 6144 wave-tiles, 8 waves/block
    qkv_proj_kernel<<<tiles / 8, 256, 0, stream>>>(xb, Wqkvb, Q, K, Vt);
  }
  relu_attn_kernel<<<BATCH * HEADS * (NSEQ / 256), 256, 0, stream>>>(Q, K, Vt, attn);
  {
    int tiles = (MTOT / 32) * (DMODEL / 64);  // 2048 wave-tiles
    out_proj_kernel<<<tiles / 8, 256, 0, stream>>>(attn, Woutb, bias, out);
  }
}
